// MultiHeadAttention_75453985456653
// MI455X (gfx1250) — compile-verified, matrix operand order checked by reading, not run on hardware
//
#include <hip/hip_runtime.h>
#include <hip/hip_bf16.h>

typedef __attribute__((ext_vector_type(16))) _Float16 v16h;
typedef __attribute__((ext_vector_type(8)))  _Float16 v8h;
typedef __attribute__((ext_vector_type(8)))  float    v8f;

#define B_   4
#define M_   8192
#define N_   512
#define D_   256
#define H_   8
#define HD_  32
#define MW_  (M_ / 32)                 /* packed mask words per row */
#define SCALE_MUL 5.656854249492381f   /* reference divides by d^-0.5 => * sqrt(32) */

// ---------- tile load helpers (A/B layout for v_wmma_f32_16x16x32_f16) ----------
// A-layout per lane: 16 halfs. lanes 0-15: k {0..7,16..23}; lanes 16-31: k {8..15,24..31}.
// B-layout is the mirror with columns on lanes. Both reduce to two 16-byte loads.

__device__ __forceinline__ v8h cvt8(const float* __restrict__ p) {
  float4 a = *(const float4*)p;
  float4 b = *(const float4*)(p + 4);
  v8h r;
  r[0]=(_Float16)a.x; r[1]=(_Float16)a.y; r[2]=(_Float16)a.z; r[3]=(_Float16)a.w;
  r[4]=(_Float16)b.x; r[5]=(_Float16)b.y; r[6]=(_Float16)b.z; r[7]=(_Float16)b.w;
  return r;
}
__device__ __forceinline__ v16h pack16(v8h lo, v8h hi) {
  v16h r;
#pragma unroll
  for (int i = 0; i < 8; ++i) { r[i] = lo[i]; r[i + 8] = hi[i]; }
  return r;
}
__device__ __forceinline__ v16h load16h(const _Float16* __restrict__ p, int off0, int off1) {
  return pack16(*(const v8h*)(p + off0), *(const v8h*)(p + off1));
}
__device__ __forceinline__ v16h load16f(const float* __restrict__ p, int off0, int off1) {
  return pack16(cvt8(p + off0), cvt8(p + off1));
}

// DPP16 max-reduction across a 16-lane half-wave (wave32: DPP rows == our halves).
__device__ __forceinline__ float rowmax16(float x) {
  int v = __float_as_int(x);
  int t = __builtin_amdgcn_update_dpp(v, v, 0xB1, 0xF, 0xF, true);   // quad_perm(1,0,3,2)
  x = fmaxf(x, __int_as_float(t));  v = __float_as_int(x);
  t = __builtin_amdgcn_update_dpp(v, v, 0x4E, 0xF, 0xF, true);       // quad_perm(2,3,0,1)
  x = fmaxf(x, __int_as_float(t));  v = __float_as_int(x);
  t = __builtin_amdgcn_update_dpp(v, v, 0x141, 0xF, 0xF, true);      // row_half_mirror
  x = fmaxf(x, __int_as_float(t));  v = __float_as_int(x);
  t = __builtin_amdgcn_update_dpp(v, v, 0x140, 0xF, 0xF, true);      // row_mirror
  x = fmaxf(x, __int_as_float(t));
  return x;
}

// CDNA5 async global->LDS copy (ASYNCcnt-tracked), inline asm = toolchain-portable.
__device__ __forceinline__ void async_ld16(void* lds_ptr, const void* gptr) {
  uint32_t l = (uint32_t)(uintptr_t)lds_ptr;   // low 32 bits of shared aperture = LDS offset
  asm volatile("global_load_async_to_lds_b128 %0, %1, off" :: "v"(l), "v"(gptr) : "memory");
}

// ---------- mask bit-packing: 32 int32 flags -> one uint32 word via ballot ----------
// Kills the 8x re-read of the 67MB mask: attention reads 2MB of words instead.
__global__ void pack_mask(const int* __restrict__ mask, unsigned* __restrict__ pm) {
  const int lane = threadIdx.x & 31;
  const int wv   = (blockIdx.x * blockDim.x + threadIdx.x) >> 5;
  const int nwv  = (gridDim.x * blockDim.x) >> 5;
  const int nwords = B_ * N_ * MW_;
  for (int w = wv; w < nwords; w += nwv) {
    const int mv = mask[(size_t)w * 32 + lane];
    const unsigned long long bal = __ballot(mv != 0);
    if (lane == 0) pm[w] = (unsigned)bal;
  }
}

// ---------- projection GEMM: out[s,o] = sum_c A[s,c] * W[o,c] + bias[o] ----------
template<bool AHALF, bool TRANS, bool OUTF32>
__global__ void proj_gemm(const void* __restrict__ Ap,
                          const float* __restrict__ W,
                          const float* __restrict__ bias,
                          void* __restrict__ outp) {
  const int lane  = threadIdx.x & 31;
  const int wave  = threadIdx.x >> 5;
  const int wt    = blockIdx.x * (blockDim.x >> 5) + wave;
  const int rt    = wt >> 4;          // row tile   (D_/16 == 16 col tiles)
  const int ot    = wt & 15;          // col tile
  const int half_ = lane >> 4;
  const int l16   = lane & 15;
  const int off0  = half_ ? 8 : 0;
  const int off1  = half_ ? 24 : 16;
  const size_t arow = (size_t)rt * 16 + l16;
  const size_t wrow = (size_t)ot * 16 + l16;

  v8f acc = {};
#pragma unroll
  for (int c0 = 0; c0 < D_; c0 += 32) {
    v16h a;
    if constexpr (AHALF)
      a = load16h((const _Float16*)Ap + arow * D_ + c0, off0, off1);
    else
      a = load16f((const float*)Ap + arow * D_ + c0, off0, off1);
    v16h b = load16f(W + wrow * D_ + c0, off0, off1);
    acc = __builtin_amdgcn_wmma_f32_16x16x32_f16(false, a, false, b, (short)0, acc, false, false);
  }
  const float bval = bias[ot * 16 + l16];
#pragma unroll
  for (int r = 0; r < 8; ++r) {         // C/D layout: VGPR r -> row r (+8 for hi half)
    const int orow = rt * 16 + r + half_ * 8;
    const int ocol = ot * 16 + l16;
    const float val = acc[r] + bval;
    if constexpr (OUTF32) {
      ((float*)outp)[(size_t)orow * D_ + ocol] = val;
    } else if constexpr (TRANS) {
      const int bidx = orow >> 13;      // / M_
      const int m    = orow & (M_ - 1);
      const int hh   = ocol >> 5, dd = ocol & 31;
      ((_Float16*)outp)[(((size_t)bidx * H_ + hh) * HD_ + dd) * M_ + m] = (_Float16)val;
    } else {
      ((_Float16*)outp)[(size_t)orow * D_ + ocol] = (_Float16)val;
    }
  }
}

// ---------- fused flash attention ----------
// Block = 4 waves, all on the same (b,h), consecutive 16-query tiles.
// K/V 32-key chunks staged to LDS with async loads, double-buffered.
#define ATTN_WAVES 4
__global__ void __launch_bounds__(128)
attn_flash(const _Float16* __restrict__ Q,    // (B*N, 256) f16
           const _Float16* __restrict__ K,    // (B*M, 256) f16
           const _Float16* __restrict__ Vt,   // (B,H,32,M) f16
           const unsigned* __restrict__ pmask,// (B,N,M/32) packed bits
           _Float16*       __restrict__ AO) { // (B*N, 256) f16
  __shared__ __align__(16) _Float16 sK[2][32 * 32];   // [key row][head dim]
  __shared__ __align__(16) _Float16 sV[2][32 * 32];   // [v dim][key offset]
  __shared__ __align__(16) _Float16 sP[ATTN_WAVES][16 * 32];

  const int tid  = threadIdx.x;
  const int lane = tid & 31;
  const int wave = tid >> 5;
  const int blk  = blockIdx.x;
  const int qb = blk & 7;               // 8 q-blocks per (b,h)
  const int h  = (blk >> 3) & 7;
  const int b  = blk >> 6;
  const int qt = qb * ATTN_WAVES + wave;

  _Float16* pbuf = sP[wave];
  const int half_ = lane >> 4, l16 = lane & 15;
  const int off0 = half_ ? 8 : 0, off1 = half_ ? 24 : 16;

  const v16h aq = load16h(Q + ((size_t)b * N_ + qt * 16 + l16) * D_ + h * HD_, off0, off1);

  const _Float16* Kb  = K  + (size_t)b * M_ * D_ + h * HD_;
  const _Float16* Vb  = Vt + ((size_t)b * H_ + h) * HD_ * M_;
  const unsigned* Mwb = pmask + (size_t)b * N_ * MW_;

  v16h onesb;
#pragma unroll
  for (int i = 0; i < 16; ++i) onesb[i] = (_Float16)1.0f;

  float mrow[8], lrow[8];
  v8f o0 = {}, o1 = {};
#pragma unroll
  for (int r = 0; r < 8; ++r) { mrow[r] = -3.0e38f; lrow[r] = 0.f; }

  const int j   = tid >> 2;             // staging: row 0..31
  const int seg = (tid & 3) * 8;        // 8-half segment within a 64B row

  // prologue: stage chunk 0
  async_ld16(&sK[0][j * 32 + seg], Kb + (size_t)j * D_ + seg);
  async_ld16(&sV[0][j * 32 + seg], Vb + (size_t)j * M_ + seg);

  const int NCH = M_ / 32;              // 256 chunks
  for (int it = 0; it < NCH; ++it) {
    const int cur = it & 1;
    const int m0  = it * 32;
    if (it + 1 < NCH) {                 // stage next chunk, then wait for current
      const int nm0 = m0 + 32;
      async_ld16(&sK[cur ^ 1][j * 32 + seg], Kb + (size_t)(nm0 + j) * D_ + seg);
      async_ld16(&sV[cur ^ 1][j * 32 + seg], Vb + (size_t)j * M_ + nm0 + seg);
      asm volatile("s_wait_asynccnt 0x2" ::: "memory");
    } else {
      asm volatile("s_wait_asynccnt 0x0" ::: "memory");
    }
    __syncthreads();

    // --- scores: K B-tiles from LDS (column n = key m0+n, contiguous 64B rows) ---
    const _Float16* kb = sK[cur];
    v16h bk0 = load16h(kb + (size_t)l16 * 32,        off0, off1);
    v16h bk1 = load16h(kb + (size_t)(16 + l16) * 32, off0, off1);
    v8f z = {};
    v8f s0 = __builtin_amdgcn_wmma_f32_16x16x32_f16(false, aq, false, bk0, (short)0, z, false, false);
    v8f s1 = __builtin_amdgcn_wmma_f32_16x16x32_f16(false, aq, false, bk1, (short)0, z, false, false);

    float p0[8], p1[8], alpha[8];
#pragma unroll
    for (int r = 0; r < 8; ++r) {
      const int q = qt * 16 + r + half_ * 8;
      const unsigned w = Mwb[(size_t)q * MW_ + it];   // one word = 32 mask bits
      const unsigned mv0 = (w >> l16) & 1u;
      const unsigned mv1 = (w >> (16 + l16)) & 1u;
      float t0 = s0[r] * SCALE_MUL;
      float t1 = s1[r] * SCALE_MUL;
      t0 = (mv0 == 0u) ? -3.0e38f : t0;
      t1 = (mv1 == 0u) ? -3.0e38f : t1;
      const float mx = rowmax16(fmaxf(t0, t1));        // DPP, no LDS pipe
      const float mn = fmaxf(mrow[r], mx);
      alpha[r] = __expf(mrow[r] - mn);
      mrow[r] = mn;
      float e0 = (mv0 == 0u) ? 0.f : __expf(t0 - mn);
      float e1 = (mv1 == 0u) ? 0.f : __expf(t1 - mn);
      o0[r] *= alpha[r];  o1[r] *= alpha[r];           // rescale running accumulator
      p0[r] = e0;  p1[r] = e1;
    }
    // --- C-layout P -> A-layout via per-wave LDS tile (same-wave LDS is in-order) ---
#pragma unroll
    for (int r = 0; r < 8; ++r) {
      const int row = r + half_ * 8;
      pbuf[row * 32 + l16]      = (_Float16)p0[r];
      pbuf[row * 32 + 16 + l16] = (_Float16)p1[r];
    }
    __builtin_amdgcn_wave_barrier();
    asm volatile("s_wait_dscnt 0x0" ::: "memory");
    const v16h ap = load16h(pbuf + l16 * 32, off0, off1);
    __builtin_amdgcn_wave_barrier();

    // --- row-sum of P on the matrix pipe: P @ ones -> every lane gets its row sum ---
    v8f csum = __builtin_amdgcn_wmma_f32_16x16x32_f16(false, ap, false, onesb, (short)0, z, false, false);
#pragma unroll
    for (int r = 0; r < 8; ++r) lrow[r] = lrow[r] * alpha[r] + csum[r];

    // --- P @ V: V B-tiles from LDS ---
    const _Float16* vb = sV[cur];
    v16h bv0 = load16h(vb + (size_t)l16 * 32,        off0, off1);
    v16h bv1 = load16h(vb + (size_t)(16 + l16) * 32, off0, off1);
    o0 = __builtin_amdgcn_wmma_f32_16x16x32_f16(false, ap, false, bv0, (short)0, o0, false, false);
    o1 = __builtin_amdgcn_wmma_f32_16x16x32_f16(false, ap, false, bv1, (short)0, o1, false, false);

    __syncthreads();                    // protect staging buffers before next refill
  }

#pragma unroll
  for (int r = 0; r < 8; ++r) {
    const float inv = (lrow[r] > 0.f) ? 1.0f / lrow[r] : 0.f;
    const int q = qt * 16 + r + half_ * 8;
    _Float16* po = AO + ((size_t)b * N_ + q) * D_ + h * HD_;
    po[l16]      = (_Float16)(o0[r] * inv);
    po[16 + l16] = (_Float16)(o1[r] * inv);
  }
}

extern "C" void kernel_launch(void* const* d_in, const int* in_sizes, int n_in,
                              void* d_out, int out_size, void* d_ws, size_t ws_size,
                              hipStream_t stream) {
  (void)in_sizes; (void)n_in; (void)out_size; (void)ws_size;
  const float* inputs_kv = (const float*)d_in[0];
  const float* inputs_q  = (const float*)d_in[1];
  const int*   mask      = (const int*)d_in[2];
  const float* Wk = (const float*)d_in[3];  const float* bk = (const float*)d_in[4];
  const float* Wq = (const float*)d_in[5];  const float* bq = (const float*)d_in[6];
  const float* Wv = (const float*)d_in[7];  const float* bv = (const float*)d_in[8];
  const float* Wo = (const float*)d_in[9];  const float* bo = (const float*)d_in[10];
  float* out = (float*)d_out;

  // workspace: Q(1MB) K(16MB) Vt(16MB) AO(1MB) f16 + packed mask (2MB) = 36MB
  _Float16* Qf = (_Float16*)d_ws;
  _Float16* Kf = Qf + (size_t)B_ * N_ * D_;
  _Float16* Vt = Kf + (size_t)B_ * M_ * D_;
  _Float16* AO = Vt + (size_t)B_ * M_ * D_;
  unsigned* PM = (unsigned*)(AO + (size_t)B_ * N_ * D_);

  const int RQ = B_ * N_;   // 2048 rows  -> RQ wave tiles
  const int RK = B_ * M_;   // 32768 rows
  dim3 blk(128);            // 4 waves/block

  pack_mask<<<2048, 256, 0, stream>>>(mask, PM);

  proj_gemm<false, false, false><<<RQ / 4, blk, 0, stream>>>(inputs_q,  Wq, bq, Qf);
  proj_gemm<false, false, false><<<RK / 4, blk, 0, stream>>>(inputs_kv, Wk, bk, Kf);
  proj_gemm<false, true,  false><<<RK / 4, blk, 0, stream>>>(inputs_kv, Wv, bv, Vt);

  attn_flash<<<B_ * H_ * 8, blk, 0, stream>>>(Qf, Kf, Vt, PM, AO);

  proj_gemm<true,  false, true ><<<RQ / 4, blk, 0, stream>>>(AO, Wo, bo, out);
}
